// HierarchicalGNNEncoder_71554155152355
// MI455X (gfx1250) — matrix operand order repaired and checked by reading, not compiled, not run on hardware
//
#include <hip/hip_runtime.h>
#include <hip/hip_bf16.h>

// ---------------------------------------------------------------------------
// GIN + virtual-node GNN forward for MI455X (gfx1250, wave32, WMMA).
// Dense MLPs -> v_wmma_f32_16x16x32_bf16 (fp32 accumulate), double-buffered
// LDS pipeline with global_load_async_to_lds_b128 (ASYNCcnt) for A tiles.
// Irregular phases -> coalesced gathers + global_atomic_add_f32 scatters.
// ---------------------------------------------------------------------------

#define N_NODES   100000
#define N_EDGES   256000
#define N_GRAPHS  4096
#define DIM       300
#define DIM2      600
#define NLAYER    5
#define AVOC      128
#define BVOC      16

// Padded GEMM dims (K multiple of 32, Ncols multiple of 64, M multiple of 128)
#define KP1       320     // 300 padded
#define KP2       608     // 600 padded
#define NP1       320     // 300 padded to col-tile
#define NP2       640     // 600 padded to col-tile
#define MP_NODES  100096  // 782 * 128
#define MP_GRAPHS 4096    // 32 * 128

typedef __attribute__((ext_vector_type(16))) __bf16 v16bf;
typedef __attribute__((ext_vector_type(8)))  __bf16 v8bf;
typedef __attribute__((ext_vector_type(8)))  float  v8f;

__device__ __forceinline__ void atomAddF(float* p, float v) {
  __hip_atomic_fetch_add(p, v, __ATOMIC_RELAXED, __HIP_MEMORY_SCOPE_AGENT);
}

// Async DMA: 16B global -> LDS, tracked by ASYNCcnt (CDNA5 ISA ch10 / §15.18.3).
// VDST supplies the per-lane LDS byte address, VADDR the 64-bit global address.
__device__ __forceinline__ void async_copy_b128(unsigned lds_off, const void* g) {
  asm volatile("global_load_async_to_lds_b128 %0, %1, off"
               :: "v"(lds_off), "v"((unsigned long long)(uintptr_t)g)
               : "memory");
}
__device__ __forceinline__ void wait_async0() {
  asm volatile("s_wait_asynccnt 0x0" ::: "memory");
}

// ---------------------------------------------------------------------------
// WMMA GEMM:  Out[M,Nc] = epilogue( A[M,Kp](bf16) @ W[Kp,Ncp](bf16) )
// epilogue: y = g[c]*(acc + b[c]) + be[c]; optional ReLU.
// Tile 128x64x32; 8 waves, each wave computes 32x32 (2x2 fragments of 16x16).
// Double-buffered LDS: A staged by async DMA, B staged+transposed via VGPRs.
// Fragment layouts follow CDNA5 ISA 7.12.2 (16-bit A 16x32, B 32x16, f32 C).
// ---------------------------------------------------------------------------
template<bool OUT_BF16>
__global__ void __launch_bounds__(256)
gin_wmma_gemm(const __bf16* __restrict__ A, int lda,
              const __bf16* __restrict__ W, int ldw,
              void* __restrict__ Out, int ldo,
              const float* __restrict__ gma, const float* __restrict__ bia,
              const float* __restrict__ bet,
              int M, int Nc, int Kp, int doRelu)
{
  // padded LDS strides (40 bf16 = 80B rows: 16B aligned segments)
  __shared__ __bf16 sA[2][128][40];   // A tiles, row-major [m][k]
  __shared__ __bf16 sB[2][64][40];    // B tiles, transposed [n][k]

  const int tid  = threadIdx.x;
  const int wave = tid >> 5;
  const int lane = tid & 31;
  const int half = lane >> 4;      // 0: lanes 0-15, 1: lanes 16-31
  const int lr   = lane & 15;

  const int rowBase = blockIdx.x * 128;
  const int colBase = blockIdx.y * 64;
  const int wr = (wave & 3) * 32;  // wave row offset
  const int wc = (wave >> 2) * 32; // wave col offset

  v8f acc[2][2] = {};

  // cooperative-load work split
  const int arow = tid >> 1;          // 0..127
  const int aseg = (tid & 1) * 16;    // 0 / 16
  const int bk   = tid >> 3;          // 0..31
  const int bc   = (tid & 7) * 8;     // 0..56

  // A tile: 128 rows x 32 K -> 2x b128 async copies per thread
  auto stageA = [&](int buf, int k0) {
    const __bf16* g = A + (long)(rowBase + arow) * lda + k0 + aseg;
    unsigned l = (unsigned)(uintptr_t)&sA[buf][arow][aseg];
    async_copy_b128(l,      g);
    async_copy_b128(l + 16, g + 8);
  };
  // B tile: W[k0..k0+31][colBase..+63] transposed into sB[buf][n][k]
  auto stageB = [&](int buf, int k0) {
    v8bf w = *(const v8bf*)(W + (long)(k0 + bk) * ldw + colBase + bc);
    #pragma unroll
    for (int j = 0; j < 8; ++j) sB[buf][bc + j][bk] = w[j];
  };

  const int nkt = Kp >> 5;
  stageA(0, 0);
  stageB(0, 0);
  wait_async0();
  __syncthreads();

  for (int kt = 0; kt < nkt; ++kt) {
    const int cur = kt & 1;
    // prefetch next K tile into the other buffer (overlaps with WMMA below)
    if (kt + 1 < nkt) {
      stageA(cur ^ 1, (kt + 1) * 32);
      stageB(cur ^ 1, (kt + 1) * 32);
    }

    // ---- fragments from current buffer ----
    union BFV { v16bf v; v8bf h[2]; };
    BFV afrag[2], bfrag[2];
    #pragma unroll
    for (int mi = 0; mi < 2; ++mi) {
      const int m = wr + mi * 16 + lr;             // A: lane holds row M=lane%16
      afrag[mi].h[0] = *(const v8bf*)&sA[cur][m][half * 8];        // K = half*8 + 0..7
      afrag[mi].h[1] = *(const v8bf*)&sA[cur][m][16 + half * 8];   // K = 16+half*8 + 0..7
    }
    #pragma unroll
    for (int ni = 0; ni < 2; ++ni) {
      const int n = wc + ni * 16 + lr;             // B: lane holds col N=lane%16
      bfrag[ni].h[0] = *(const v8bf*)&sB[cur][n][half * 8];        // K = half*8 + 0..7
      bfrag[ni].h[1] = *(const v8bf*)&sB[cur][n][16 + half * 8];   // K = 16+half*8 + 0..7
    }
    #pragma unroll
    for (int mi = 0; mi < 2; ++mi)
      #pragma unroll
      for (int ni = 0; ni < 2; ++ni)
        acc[mi][ni] = __builtin_amdgcn_wmma_f32_16x16x32_bf16(
            false, afrag[mi].v, false, bfrag[ni].v,
            (short)0, acc[mi][ni], false, false);

    // own async copies (next buffer) must have landed before the barrier;
    // the DMA had the whole compute phase to complete, so this is cheap.
    wait_async0();
    __syncthreads();
  }

  // ---- epilogue: C layout: elem j -> row = half*8+j, col = lane%16 ----
  #pragma unroll
  for (int ni = 0; ni < 2; ++ni) {
    const int col = colBase + wc + ni * 16 + lr;
    float g = 0.f, b = 0.f, be = 0.f;
    if (col < Nc) { g = gma[col]; b = bia[col]; be = bet[col]; }
    #pragma unroll
    for (int mi = 0; mi < 2; ++mi) {
      #pragma unroll
      for (int j = 0; j < 8; ++j) {
        const int row = rowBase + wr + mi * 16 + half * 8 + j;
        if (row < M && col < Nc) {
          float y = g * (acc[mi][ni][j] + b) + be;
          if (doRelu) y = fmaxf(y, 0.f);
          if (OUT_BF16) ((__bf16*)Out)[(long)row * ldo + col] = (__bf16)y;
          else          ((float*)Out)[(long)row * ldo + col]  = y;
        }
      }
    }
  }
}

// ---------------------------------------------------------------------------
// Elementwise / irregular kernels
// ---------------------------------------------------------------------------
__global__ void fill_zero_u32(unsigned int* p, long n) {
  long i = (long)blockIdx.x * blockDim.x + threadIdx.x;
  if (i < n) p[i] = 0u;
}

// h[n,:] = sum_c atom_emb[c, x[n,c], :]
__global__ void __launch_bounds__(128)
atom_encoder(const float* __restrict__ atom_emb, const int* __restrict__ x,
             float* __restrict__ h)
{
  const int n = blockIdx.x;
  int idx[9];
  #pragma unroll
  for (int c = 0; c < 9; ++c) idx[c] = x[n * 9 + c];
  for (int d = threadIdx.x; d < DIM; d += 128) {
    float s = 0.f;
    #pragma unroll
    for (int c = 0; c < 9; ++c) s += atom_emb[((long)c * AVOC + idx[c]) * DIM + d];
    h[(long)n * DIM + d] = s;
  }
}

__global__ void vn_broadcast(const float* __restrict__ vemb, float* __restrict__ vn, long total) {
  long i = (long)blockIdx.x * blockDim.x + threadIdx.x;
  if (i < total) vn[i] = vemb[i % DIM];
}

__global__ void add_vn(const float* __restrict__ h, const float* __restrict__ vn,
                       const int* __restrict__ batch, float* __restrict__ h_in, long total) {
  long i = (long)blockIdx.x * blockDim.x + threadIdx.x;
  if (i >= total) return;
  int n = (int)(i / DIM);
  int d = (int)(i - (long)n * DIM);
  h_in[i] = h[i] + vn[(long)batch[n] * DIM + d];
}

// agg[dst] += relu(h_in[src] + bond_embed(edge_attr))
__global__ void __launch_bounds__(128)
edge_message(const float* __restrict__ h_in, const float* __restrict__ bond_l,
             const int* __restrict__ eattr, const int* __restrict__ esrc,
             const int* __restrict__ edst, float* __restrict__ agg)
{
  const int e = blockIdx.x;
  const int s = esrc[e], t = edst[e];
  const int a0 = eattr[e * 3 + 0], a1 = eattr[e * 3 + 1], a2 = eattr[e * 3 + 2];
  const float* b0 = bond_l + ((long)0 * BVOC + a0) * DIM;
  const float* b1 = bond_l + ((long)1 * BVOC + a1) * DIM;
  const float* b2 = bond_l + ((long)2 * BVOC + a2) * DIM;
  const float* hs = h_in + (long)s * DIM;
  float* ag = agg + (long)t * DIM;
  for (int d = threadIdx.x; d < DIM; d += 128) {
    float m = hs[d] + b0[d] + b1[d] + b2[d];
    m = fmaxf(m, 0.f);
    atomAddF(&ag[d], m);
  }
}

// zA (bf16, [Mp x Kp], zero padded) = (1+eps)*h_in + agg
__global__ void combine_z(const float* __restrict__ h_in, const float* __restrict__ agg,
                          const float* __restrict__ eps_l, __bf16* __restrict__ out,
                          int M, int K, int Kp, long total) {
  long i = (long)blockIdx.x * blockDim.x + threadIdx.x;
  if (i >= total) return;
  int r = (int)(i / Kp);
  int k = (int)(i - (long)r * Kp);
  float v = 0.f;
  if (r < M && k < K) {
    long src = (long)r * K + k;
    v = (1.0f + eps_l[0]) * h_in[src] + agg[src];
  }
  out[i] = (__bf16)v;
}

// out (bf16, [Rp x Cp], zero padded) = in (f32, [R x C])
__global__ void convert_pad(const float* __restrict__ in, __bf16* __restrict__ out,
                            int R, int C, int Cp, long total) {
  long i = (long)blockIdx.x * blockDim.x + threadIdx.x;
  if (i >= total) return;
  int r = (int)(i / Cp);
  int c = (int)(i - (long)r * Cp);
  float v = (r < R && c < C) ? in[(long)r * C + c] : 0.f;
  out[i] = (__bf16)v;
}

__global__ void copy_f32(const float* __restrict__ a, float* __restrict__ b, long n) {
  long i = (long)blockIdx.x * blockDim.x + threadIdx.x;
  if (i < n) b[i] = a[i];
}

// dst[seg[n], d] += src[n, d]
__global__ void scatter_add_rows(const float* __restrict__ src, const int* __restrict__ seg,
                                 float* __restrict__ dst, long total, int C) {
  long i = (long)blockIdx.x * blockDim.x + threadIdx.x;
  if (i >= total) return;
  int n = (int)(i / C);
  int d = (int)(i - (long)n * C);
  atomAddF(&dst[(long)seg[n] * C + d], src[i]);
}

__global__ void count_nodes(const int* __restrict__ batch, float* __restrict__ counts, int n) {
  int i = blockIdx.x * blockDim.x + threadIdx.x;
  if (i < n) atomAddF(&counts[batch[i]], 1.0f);
}

__global__ void finalize_mean(float* __restrict__ out, const float* __restrict__ counts,
                              long total, int C) {
  long i = (long)blockIdx.x * blockDim.x + threadIdx.x;
  if (i >= total) return;
  int b = (int)(i / C);
  out[i] /= fmaxf(counts[b], 1.0f);
}

// ---------------------------------------------------------------------------
// Host orchestration
// ---------------------------------------------------------------------------
static inline long cdiv(long a, long b) { return (a + b - 1) / b; }

extern "C" void kernel_launch(void* const* d_in, const int* in_sizes, int n_in,
                              void* d_out, int out_size, void* d_ws, size_t ws_size,
                              hipStream_t stream) {
  (void)in_sizes; (void)n_in; (void)out_size; (void)ws_size;
  const float* atom_emb = (const float*)d_in[0];
  const float* bond_emb = (const float*)d_in[1];
  const float* eps      = (const float*)d_in[2];
  const float* W1   = (const float*)d_in[3];
  const float* b1   = (const float*)d_in[4];
  const float* g1   = (const float*)d_in[5];
  const float* be1  = (const float*)d_in[6];
  const float* W2   = (const float*)d_in[7];
  const float* b2   = (const float*)d_in[8];
  const float* g_bn = (const float*)d_in[9];
  const float* b_bn = (const float*)d_in[10];
  const float* vn_emb = (const float*)d_in[11];
  const float* Wv1  = (const float*)d_in[12];
  const float* bv1  = (const float*)d_in[13];
  const float* gv1  = (const float*)d_in[14];
  const float* bev1 = (const float*)d_in[15];
  const float* Wv2  = (const float*)d_in[16];
  const float* bv2  = (const float*)d_in[17];
  const float* gv2  = (const float*)d_in[18];
  const float* bev2 = (const float*)d_in[19];
  const int* x         = (const int*)d_in[20];
  const int* edge_attr = (const int*)d_in[21];
  const int* edge_index= (const int*)d_in[22];
  const int* batch     = (const int*)d_in[23];
  const int* esrc = edge_index;
  const int* edst = edge_index + N_EDGES;
  float* out = (float*)d_out;

  // ---- workspace carve-up ----
  char* wsp = (char*)d_ws;
  auto alloc = [&](size_t bytes) { char* p = wsp; wsp += (bytes + 255) & ~(size_t)255; return p; };
  float*  h      = (float*)alloc((size_t)N_NODES * DIM * 4);
  float*  h_in   = (float*)alloc((size_t)N_NODES * DIM * 4);
  float*  agg    = (float*)alloc((size_t)N_NODES * DIM * 4);
  float*  vn     = (float*)alloc((size_t)N_GRAPHS * DIM * 4);
  float*  vn_tmp = (float*)alloc((size_t)N_GRAPHS * DIM * 4);
  __bf16* zA     = (__bf16*)alloc((size_t)MP_NODES * KP1 * 2);
  __bf16* zMid   = (__bf16*)alloc((size_t)MP_NODES * KP2 * 2);
  __bf16* vA     = (__bf16*)alloc((size_t)MP_GRAPHS * KP1 * 2);
  __bf16* vMid   = (__bf16*)alloc((size_t)MP_GRAPHS * KP2 * 2);
  __bf16* w1b    = (__bf16*)alloc((size_t)NLAYER * KP1 * NP2 * 2);
  __bf16* w2b    = (__bf16*)alloc((size_t)NLAYER * KP2 * NP1 * 2);
  __bf16* wv1b   = (__bf16*)alloc((size_t)(NLAYER - 1) * KP1 * NP2 * 2);
  __bf16* wv2b   = (__bf16*)alloc((size_t)(NLAYER - 1) * KP2 * NP1 * 2);
  float*  counts = (float*)alloc((size_t)N_GRAPHS * 4);

  const int TB = 256;
  auto zero = [&](void* p, long words) {
    fill_zero_u32<<<dim3((unsigned)cdiv(words, TB)), dim3(TB), 0, stream>>>((unsigned int*)p, words);
  };

  // zero pad-sensitive buffers + accumulators (ws/d_out are poisoned by harness)
  zero(zMid, (long)MP_NODES * KP2 / 2);
  zero(vMid, (long)MP_GRAPHS * KP2 / 2);
  zero(counts, N_GRAPHS);
  zero(out, (long)N_GRAPHS * DIM);

  // convert weights -> bf16, zero-padded [Kp x Ncp]
  for (int l = 0; l < NLAYER; ++l) {
    long t1 = (long)KP1 * NP2;
    convert_pad<<<dim3((unsigned)cdiv(t1, TB)), dim3(TB), 0, stream>>>(
        W1 + (long)l * DIM * DIM2, w1b + (long)l * KP1 * NP2, DIM, DIM2, NP2, t1);
    long t2 = (long)KP2 * NP1;
    convert_pad<<<dim3((unsigned)cdiv(t2, TB)), dim3(TB), 0, stream>>>(
        W2 + (long)l * DIM2 * DIM, w2b + (long)l * KP2 * NP1, DIM2, DIM, NP1, t2);
  }
  for (int l = 0; l < NLAYER - 1; ++l) {
    long t1 = (long)KP1 * NP2;
    convert_pad<<<dim3((unsigned)cdiv(t1, TB)), dim3(TB), 0, stream>>>(
        Wv1 + (long)l * DIM * DIM2, wv1b + (long)l * KP1 * NP2, DIM, DIM2, NP2, t1);
    long t2 = (long)KP2 * NP1;
    convert_pad<<<dim3((unsigned)cdiv(t2, TB)), dim3(TB), 0, stream>>>(
        Wv2 + (long)l * DIM2 * DIM, wv2b + (long)l * KP2 * NP1, DIM2, DIM, NP1, t2);
  }

  // encoders / init
  atom_encoder<<<dim3(N_NODES), dim3(128), 0, stream>>>(atom_emb, x, h);
  vn_broadcast<<<dim3((unsigned)cdiv((long)N_GRAPHS * DIM, TB)), dim3(TB), 0, stream>>>(
      vn_emb, vn, (long)N_GRAPHS * DIM);

  const long ND = (long)N_NODES * DIM;
  const long BD = (long)N_GRAPHS * DIM;

  for (int l = 0; l < NLAYER; ++l) {
    // h_in = h + vn[batch]
    add_vn<<<dim3((unsigned)cdiv(ND, TB)), dim3(TB), 0, stream>>>(h, vn, batch, h_in, ND);
    // agg = scatter_add(relu(h_in[src] + bond_e))
    zero(agg, ND);
    edge_message<<<dim3(N_EDGES), dim3(128), 0, stream>>>(
        h_in, bond_emb + (long)l * 3 * BVOC * DIM, edge_attr, esrc, edst, agg);
    // zA = bf16((1+eps)*h_in + agg), padded
    long tz = (long)MP_NODES * KP1;
    combine_z<<<dim3((unsigned)cdiv(tz, TB)), dim3(TB), 0, stream>>>(
        h_in, agg, eps + l, zA, N_NODES, DIM, KP1, tz);
    // zMid = relu(g1*(zA@W1 + b1) + be1)   [100000 x 600] bf16
    gin_wmma_gemm<true><<<dim3(MP_NODES / 128, NP2 / 64), dim3(256), 0, stream>>>(
        zA, KP1, w1b + (long)l * KP1 * NP2, NP2, zMid, KP2,
        g1 + (long)l * DIM2, b1 + (long)l * DIM2, be1 + (long)l * DIM2,
        N_NODES, DIM2, KP1, 1);
    // h = g_bn*(zMid@W2 + b2) + b_bn  (+relu if l<L-1)  [100000 x 300] f32
    gin_wmma_gemm<false><<<dim3(MP_NODES / 128, NP1 / 64), dim3(256), 0, stream>>>(
        zMid, KP2, w2b + (long)l * KP2 * NP1, NP1, h, DIM,
        g_bn + (long)l * DIM, b2 + (long)l * DIM, b_bn + (long)l * DIM,
        N_NODES, DIM, KP2, (l < NLAYER - 1) ? 1 : 0);

    if (l < NLAYER - 1) {
      // vn_tmp = segment_sum(h_in, batch) + vn
      copy_f32<<<dim3((unsigned)cdiv(BD, TB)), dim3(TB), 0, stream>>>(vn, vn_tmp, BD);
      scatter_add_rows<<<dim3((unsigned)cdiv(ND, TB)), dim3(TB), 0, stream>>>(
          h_in, batch, vn_tmp, ND, DIM);
      long tv = (long)MP_GRAPHS * KP1;
      convert_pad<<<dim3((unsigned)cdiv(tv, TB)), dim3(TB), 0, stream>>>(
          vn_tmp, vA, N_GRAPHS, DIM, KP1, tv);
      // vMid = relu(gv1*(vA@Wv1 + bv1) + bev1)
      gin_wmma_gemm<true><<<dim3(MP_GRAPHS / 128, NP2 / 64), dim3(256), 0, stream>>>(
          vA, KP1, wv1b + (long)l * KP1 * NP2, NP2, vMid, KP2,
          gv1 + (long)l * DIM2, bv1 + (long)l * DIM2, bev1 + (long)l * DIM2,
          N_GRAPHS, DIM2, KP1, 1);
      // vn = relu(gv2*(vMid@Wv2 + bv2) + bev2)
      gin_wmma_gemm<false><<<dim3(MP_GRAPHS / 128, NP1 / 64), dim3(256), 0, stream>>>(
          vMid, KP2, wv2b + (long)l * KP2 * NP1, NP1, vn, DIM,
          gv2 + (long)l * DIM, bv2 + (long)l * DIM, bev2 + (long)l * DIM,
          N_GRAPHS, DIM, KP2, 1);
    }
  }

  // global mean pool
  count_nodes<<<dim3((unsigned)cdiv(N_NODES, TB)), dim3(TB), 0, stream>>>(batch, counts, N_NODES);
  scatter_add_rows<<<dim3((unsigned)cdiv(ND, TB)), dim3(TB), 0, stream>>>(h, batch, out, ND, DIM);
  finalize_mean<<<dim3((unsigned)cdiv(BD, TB)), dim3(TB), 0, stream>>>(out, counts, BD, DIM);
}
